// DCNv2_42417097016101
// MI455X (gfx1250) — matrix operand order repaired and checked by reading, not compiled, not run on hardware
//
#include <hip/hip_runtime.h>

typedef unsigned short u16;
typedef __attribute__((ext_vector_type(16))) __bf16 v16bf;
typedef __attribute__((ext_vector_type(8)))  float  v8f;

#define Bk   8
#define Ck   256
#define Hk   64
#define Wk   64
#define Ok   256
#define KKk  9
#define HWk  4096          // H*W = Ho*Wo plane
#define KTOT 72            // 9 taps * 8 channel-chunks of 32
#define APAD 40            // bf16 elements per LDS tile row (32 data + 8 pad, keeps 16B align)

// ---------------------------------------------------------------------------
// Pre-kernel: weight [O, C, 3, 3] fp32  ->  Wb[k][o][c] bf16 (c contiguous)
// ---------------------------------------------------------------------------
__global__ void dcn_weight_to_bf16(const float* __restrict__ w, u16* __restrict__ wb) {
    int e = blockIdx.x * 256 + threadIdx.x;          // e = (k*256 + o)*256 + c
    if (e >= KKk * Ok * Ck) return;
    int k = e >> 16;
    int rem = e & 65535;
    int o = rem >> 8;
    int c = rem & 255;
    float f = w[(o * Ck + c) * KKk + k];
    union { __bf16 h; u16 u; } cv;
    cv.h = (__bf16)f;
    wb[e] = cv.u;
}

// ---------------------------------------------------------------------------
// Per-thread bilinear coordinate cache (valid for one tap k, one pixel n)
// ---------------------------------------------------------------------------
struct Coords {
    int   x0, x1, rT, rB;          // clamped column indices and row*W offsets
    float w00, w01, w10, w11;      // corner weights, pre-multiplied by mask, zeroed if OOB
};

__device__ __forceinline__ void compute_coords(const float* __restrict__ off,
                                               const float* __restrict__ msk,
                                               int b, int h, int n, int k, Coords& cd) {
    int hw = h * Wk + n;
    float oy = off[((b * 2 * KKk + 2 * k)     << 12) + hw];
    float ox = off[((b * 2 * KKk + 2 * k + 1) << 12) + hw];
    float m  = msk[((b * KKk + k)             << 12) + hw];
    int ki = k / 3, kj = k - ki * 3;
    float py = (float)(h + ki - 1) + oy;
    float px = (float)(n + kj - 1) + ox;
    float fy = floorf(py), fx = floorf(px);
    float ly = py - fy,    lx = px - fx;
    int y0 = (int)fy, x0 = (int)fx;
    bool vy0 = (y0 >= 0)  && (y0 < Hk);
    bool vy1 = (y0 >= -1) && (y0 < Hk - 1);
    bool vx0 = (x0 >= 0)  && (x0 < Wk);
    bool vx1 = (x0 >= -1) && (x0 < Wk - 1);
    float hy = 1.0f - ly, hx = 1.0f - lx;
    cd.w00 = (vy0 && vx0) ? hy * hx * m : 0.0f;
    cd.w01 = (vy0 && vx1) ? hy * lx * m : 0.0f;
    cd.w10 = (vy1 && vx0) ? ly * hx * m : 0.0f;
    cd.w11 = (vy1 && vx1) ? ly * lx * m : 0.0f;
    int y0c = min(max(y0, 0), Hk - 1);
    int y1c = min(max(y0 + 1, 0), Hk - 1);
    cd.x0 = min(max(x0, 0), Wk - 1);
    cd.x1 = min(max(x0 + 1, 0), Wk - 1);
    cd.rT = y0c * Wk;
    cd.rB = y1c * Wk;
}

// ---------------------------------------------------------------------------
// Build one K-chunk: sA[256 o][32 c] from Wb, sB[64 n][32 c] via bilinear im2col
// ---------------------------------------------------------------------------
__device__ __forceinline__ void build_tile(u16* __restrict__ sAbuf, u16* __restrict__ sBbuf,
                                           const u16* __restrict__ wb,
                                           const float* __restrict__ inp,
                                           const float* __restrict__ off,
                                           const float* __restrict__ msk,
                                           int b, int h, int kit, int tid, Coords& cd) {
    int k     = kit >> 3;
    int cbase = (kit & 7) << 5;

    // refresh bilinear coords when the kernel tap changes (thread-private)
    if ((kit & 7) == 0)
        compute_coords(off, msk, b, h, tid & 63, k, cd);

    // --- A tile: 64 contiguous bytes of pre-converted bf16 weights per row ---
    {
        int o = tid;
        const uint4* src = (const uint4*)(wb + ((k << 8) + o) * Ck + cbase);
        uint4* dst = (uint4*)(sAbuf + o * APAD);
        dst[0] = src[0]; dst[1] = src[1]; dst[2] = src[2]; dst[3] = src[3];
    }

    // --- B tile: thread (n = tid&63, cq = tid>>6) samples 8 channels ---
    {
        int n  = tid & 63;
        int cq = tid >> 6;
        const float* img = inp + ((size_t)b * Ck + cbase + cq * 8) * HWk;
        union { __bf16 h8[8]; uint4 u; } pk;
#pragma unroll
        for (int ci = 0; ci < 8; ++ci) {
            const float* p = img + ci * HWk;
            float v00 = p[cd.rT + cd.x0];
            float v01 = p[cd.rT + cd.x1];
            float v10 = p[cd.rB + cd.x0];
            float v11 = p[cd.rB + cd.x1];
            float val = v00 * cd.w00 + v01 * cd.w01 + v10 * cd.w10 + v11 * cd.w11;
            pk.h8[ci] = (__bf16)val;
        }
        *(uint4*)(sBbuf + n * APAD + cq * 8) = pk.u;
    }
}

// ---------------------------------------------------------------------------
// Consume one K-chunk: 2 A-frags x 4 B-frags -> 8 WMMAs per wave
// ---------------------------------------------------------------------------
__device__ __forceinline__ void compute_tile(const u16* __restrict__ sAbuf,
                                             const u16* __restrict__ sBbuf,
                                             int wave, int lane, v8f acc[2][4]) {
    union FragU { uint4 u[2]; v16bf v; };
    int row  = lane & 15;
    int half = lane >> 4;

    FragU afr[2];
#pragma unroll
    for (int mi = 0; mi < 2; ++mi) {
        // A 16x32 operand, contiguous k order chosen here (same as B): lane holds K = half*16..+15 of row M=lane%16
        const u16* pa = sAbuf + (wave * 32 + mi * 16 + row) * APAD + half * 16;
        afr[mi].u[0] = *(const uint4*)pa;          // K: half*16 .. half*16+7
        afr[mi].u[1] = *(const uint4*)(pa + 8);    // K: half*16+8 .. +7
    }

    FragU bfr[4];
#pragma unroll
    for (int j = 0; j < 4; ++j) {
        // B 32x16 operand, contiguous k order chosen here (same as A): lane holds K = half*16..+15 of column N=lane%16
        const u16* pb = sBbuf + (j * 16 + row) * APAD + half * 16;
        bfr[j].u[0] = *(const uint4*)pb;
        bfr[j].u[1] = *(const uint4*)(pb + 8);
    }

#pragma unroll
    for (int mi = 0; mi < 2; ++mi)
#pragma unroll
        for (int j = 0; j < 4; ++j)
            acc[mi][j] = __builtin_amdgcn_wmma_f32_16x16x32_bf16(
                false, afr[mi].v, false, bfr[j].v,
                (short)0, acc[mi][j], false, false);
}

// ---------------------------------------------------------------------------
// Fused DCNv2 forward: one block per (b, output row h); tile M=256 x N=64
// ---------------------------------------------------------------------------
__global__ __launch_bounds__(256) void dcn_wmma_kernel(
        const float* __restrict__ inp, const float* __restrict__ off,
        const float* __restrict__ msk, const float* __restrict__ bias,
        const u16* __restrict__ wb, float* __restrict__ out) {

    __shared__ u16  sA[2][Ok * APAD];      // 2 x 20 KB
    __shared__ u16  sB[2][64 * APAD];      // 2 x 5 KB
    __shared__ float sBias[Ok];

    int tid  = threadIdx.x;
    int b    = blockIdx.x >> 6;
    int h    = blockIdx.x & 63;
    int lane = tid & 31;
    int wave = tid >> 5;

    sBias[tid] = bias[tid];

    v8f acc[2][4];
    v8f z = {0.f, 0.f, 0.f, 0.f, 0.f, 0.f, 0.f, 0.f};
#pragma unroll
    for (int mi = 0; mi < 2; ++mi)
#pragma unroll
        for (int j = 0; j < 4; ++j)
            acc[mi][j] = z;

    Coords cd;
    build_tile(sA[0], sB[0], wb, inp, off, msk, b, h, 0, tid, cd);
    __syncthreads();

    for (int kit = 0; kit < KTOT; ++kit) {
        int cur = kit & 1;
        if (kit + 1 < KTOT)
            build_tile(sA[cur ^ 1], sB[cur ^ 1], wb, inp, off, msk, b, h, kit + 1, tid, cd);
        compute_tile(sA[cur], sB[cur], wave, lane, acc);
        __syncthreads();
    }

    // Epilogue: C/D 16x16 f32 layout -> VGPR r holds M=r (lanes 0-15) / M=8+r (lanes 16-31)
    int msub = (lane >> 4) * 8;
    int ncol = lane & 15;
#pragma unroll
    for (int mi = 0; mi < 2; ++mi)
#pragma unroll
        for (int j = 0; j < 4; ++j) {
            int n = j * 16 + ncol;
#pragma unroll
            for (int r = 0; r < 8; ++r) {
                int m = wave * 32 + mi * 16 + msub + r;
                out[(((size_t)b * Ok + m) << 12) + h * Wk + n] = acc[mi][j][r] + sBias[m];
            }
        }
}

// ---------------------------------------------------------------------------
extern "C" void kernel_launch(void* const* d_in, const int* in_sizes, int n_in,
                              void* d_out, int out_size, void* d_ws, size_t ws_size,
                              hipStream_t stream) {
    const float* inp  = (const float*)d_in[0];   // [8,256,64,64]
    const float* off  = (const float*)d_in[1];   // [8,18,64,64]
    const float* msk  = (const float*)d_in[2];   // [8,9,64,64]
    const float* wgt  = (const float*)d_in[3];   // [256,256,3,3]
    const float* bias = (const float*)d_in[4];   // [256]
    float* out = (float*)d_out;

    u16* wb = (u16*)d_ws;                        // needs 9*256*256*2 = 1,179,648 B
    (void)in_sizes; (void)n_in; (void)out_size; (void)ws_size;

    dcn_weight_to_bf16<<<(KKk * Ok * Ck + 255) / 256, 256, 0, stream>>>(wgt, wb);
    dcn_wmma_kernel<<<Bk * Hk, 256, 0, stream>>>(inp, off, msk, bias, wb, out);
}
